// EncoderNet_16475494547643
// MI455X (gfx1250) — compile-verified, matrix operand order checked by reading, not run on hardware
//
#include <hip/hip_runtime.h>
#include <hip/hip_bf16.h>
#include <math.h>

// ---------------------------------------------------------------------------
// GRU encoder-decoder for MI455X (gfx1250, wave32, WMMA).
// All matrix math via v_wmma_f32_16x16x32_bf16 (bf16 in, f32 accumulate),
// gate math / hidden-state master copy in f32. GEMM inner loops use a
// copy-free ping-pong (K unrolled by 2, two register fragment sets that swap
// roles), so WMMAs overlap the in-flight loads of the next K-tile with no
// v_mov rotation and only partial s_wait_loadcnt -- critical for the
// 255-step serial scan which runs with only 32 blocks.
// ---------------------------------------------------------------------------

typedef __bf16 bf16_t;
typedef __attribute__((ext_vector_type(8)))  bf16_t v8bf;
typedef __attribute__((ext_vector_type(16))) bf16_t v16bf;
typedef __attribute__((ext_vector_type(8)))  float  v8f;

#define TM1   255              // T-1 timesteps actually computed
#define BB    128              // batch
#define II    512              // input dim
#define HH    1024             // hidden dim
#define OO    256              // output dim
#define GG    (3 * HH)         // 3072 gate columns
#define MROWS (TM1 * BB)       // 32640 flattened (t,b) rows

// Load a 16x32 bf16 WMMA fragment from a row-major [rows, ldk] matrix.
// CDNA5 16-bit A/B fragment layout: lane L covers row (L%16); VGPRs 0-3 hold
// K = (L/16)*8 .. +7, VGPRs 4-7 hold K = 16+(L/16)*8 .. +7 (2 bf16/VGPR).
// Both A ([M,K] row-major) and B (= W^T with W [N,K] row-major) use this.
__device__ __forceinline__ v16bf load_frag_rowmajor(const bf16_t* __restrict__ base,
                                                    int r0, int ldk, int k0, int lane) {
    const int row = r0 + (lane & 15);
    const int kh  = (lane >> 4) << 3;            // 0 or 8
    const bf16_t* p = base + (size_t)row * ldk + (size_t)(k0 + kh);
    v8bf lo = *(const v8bf*)p;                   // K = k0+kh    .. +7  (16B)
    v8bf hi = *(const v8bf*)(p + 16);            // K = k0+16+kh .. +7  (16B)
    return __builtin_shufflevector(lo, hi, 0, 1, 2, 3, 4, 5, 6, 7,
                                           8, 9, 10, 11, 12, 13, 14, 15);
}

// Copy-free software-pipelined GEMM core: accumulates NKT K-tiles (32 each)
// into c[NB] for one 16-row A tile and NB 16-col B tiles. NKT must be even
// and >= 4. Two fragment sets alternate roles inside one loop iteration, so
// no register rotation is needed and each WMMA group runs with exactly one
// fragment-set of global loads still in flight.
template <int NB, int NKT>
__device__ __forceinline__ void wmma_gemm_pipe(const bf16_t* __restrict__ A, int am0, int lda,
                                               const bf16_t* __restrict__ B,
                                               const int (&brow)[NB], int ldb,
                                               int lane, v8f (&c)[NB]) {
    v16bf a0, a1, b0[NB], b1[NB];
    a0 = load_frag_rowmajor(A, am0, lda, 0, lane);
#pragma unroll
    for (int i = 0; i < NB; ++i) b0[i] = load_frag_rowmajor(B, brow[i], ldb, 0, lane);

#pragma unroll 1
    for (int it = 0; it < NKT / 2 - 1; ++it) {
        const int ka = (2 * it + 1) * 32;
        const int kb = (2 * it + 2) * 32;
        // issue loads for tile ka into set1, then consume set0 (tile 2*it)
        a1 = load_frag_rowmajor(A, am0, lda, ka, lane);
#pragma unroll
        for (int i = 0; i < NB; ++i) b1[i] = load_frag_rowmajor(B, brow[i], ldb, ka, lane);
#pragma unroll
        for (int i = 0; i < NB; ++i)
            c[i] = __builtin_amdgcn_wmma_f32_16x16x32_bf16(false, a0, false, b0[i],
                                                           (short)0, c[i], false, false);
        // issue loads for tile kb into set0, then consume set1 (tile ka)
        a0 = load_frag_rowmajor(A, am0, lda, kb, lane);
#pragma unroll
        for (int i = 0; i < NB; ++i) b0[i] = load_frag_rowmajor(B, brow[i], ldb, kb, lane);
#pragma unroll
        for (int i = 0; i < NB; ++i)
            c[i] = __builtin_amdgcn_wmma_f32_16x16x32_bf16(false, a1, false, b1[i],
                                                           (short)0, c[i], false, false);
    }
    // Tail: set0 holds tile NKT-2; load tile NKT-1 into set1, consume both.
    {
        const int klast = (NKT - 1) * 32;
        a1 = load_frag_rowmajor(A, am0, lda, klast, lane);
#pragma unroll
        for (int i = 0; i < NB; ++i) b1[i] = load_frag_rowmajor(B, brow[i], ldb, klast, lane);
#pragma unroll
        for (int i = 0; i < NB; ++i)
            c[i] = __builtin_amdgcn_wmma_f32_16x16x32_bf16(false, a0, false, b0[i],
                                                           (short)0, c[i], false, false);
#pragma unroll
        for (int i = 0; i < NB; ++i)
            c[i] = __builtin_amdgcn_wmma_f32_16x16x32_bf16(false, a1, false, b1[i],
                                                           (short)0, c[i], false, false);
    }
}

__device__ __forceinline__ float sigmoidf_(float x) {
    return 1.0f / (1.0f + __expf(-x));
}

// ---------------------------------------------------------------------------
// f32 -> bf16 conversion (grid-strided)
// ---------------------------------------------------------------------------
__global__ void __launch_bounds__(256) cvt_f32_bf16_kernel(const float* __restrict__ src,
                                                           bf16_t* __restrict__ dst, int n) {
    int i = blockIdx.x * blockDim.x + threadIdx.x;
    const int stride = gridDim.x * blockDim.x;
    for (; i < n; i += stride) dst[i] = (bf16_t)src[i];
}

// Zero h0 (f32 + bf16 ping-pong slot 0) and the loss accumulator d_out[0].
__global__ void __launch_bounds__(256) init_kernel(float* __restrict__ hf0,
                                                   bf16_t* __restrict__ hb0,
                                                   float* __restrict__ out0) {
    const int i = blockIdx.x * blockDim.x + threadIdx.x;
    if (i < BB * HH) { hf0[i] = 0.0f; hb0[i] = (bf16_t)0.0f; }
    if (i == 0) *out0 = 0.0f;
}

// ---------------------------------------------------------------------------
// gi = x[:255] @ W_ih^T + b_ih     ([32640,512] x [3072,512]^T -> [32640,3072])
// grid (255, 48), block 256 (8 waves). Wave w: rows [bx*128 + w*16, +16),
// cols [by*64, +64) as 4 accumulators.
// ---------------------------------------------------------------------------
__global__ void __launch_bounds__(256) gi_gemm_kernel(const bf16_t* __restrict__ xb,
                                                      const bf16_t* __restrict__ wih,
                                                      const float*  __restrict__ bih,
                                                      float* __restrict__ gi) {
    const int lane = threadIdx.x & 31;
    const int wid  = threadIdx.x >> 5;
    const int m0 = blockIdx.x * 128 + wid * 16;
    const int n0 = blockIdx.y * 64;

    v8f c[4] = {};
    const int brow[4] = { n0, n0 + 16, n0 + 32, n0 + 48 };
    wmma_gemm_pipe<4, II / 32>(xb, m0, II, wih, brow, II, lane, c);

    // Epilogue: lane L holds C[m = v + 8*(L/16), n = L%16] in c[nt][v].
    const int rbase = m0 + ((lane >> 4) << 3);
#pragma unroll
    for (int nt = 0; nt < 4; ++nt) {
        const int col = n0 + nt * 16 + (lane & 15);
        const float bias = bih[col];
#pragma unroll
        for (int v = 0; v < 8; ++v) {
            gi[(size_t)(rbase + v) * GG + col] = c[nt][v] + bias;
        }
    }
}

// ---------------------------------------------------------------------------
// One GRU step, fully fused: gh = h @ W_hh^T for all 3 gates, then gates +
// hidden-state update. grid = 32 blocks (32 hidden cols each), block 256.
// Wave w: rows [w*16,+16) x 3 gates x 2 col-tiles = 6 accumulators (48 VGPRs);
// ping-pong A (2x8) + B (2x48) fragments stay under the VGPR budget.
// Ping-pong h buffers: reads *_old, writes *_new (disjoint cols per block).
// ---------------------------------------------------------------------------
__global__ void __launch_bounds__(256) gru_step_kernel(const bf16_t* __restrict__ hb_old,
                                                       const float*  __restrict__ hf_old,
                                                       bf16_t* __restrict__ hb_new,
                                                       float*  __restrict__ hf_new,
                                                       const bf16_t* __restrict__ whh,   // [3072,1024] bf16
                                                       const float*  __restrict__ bhh,   // [3072]
                                                       const float*  __restrict__ gi_t,  // [128,3072] (incl b_ih)
                                                       bf16_t* __restrict__ hs_t) {      // [128,1024] bf16 out
    const int lane = threadIdx.x & 31;
    const int wid  = threadIdx.x >> 5;
    const int m0 = wid * 16;               // 8 waves cover all 128 batch rows
    const int j0 = blockIdx.x * 32;        // hidden-column tile (32 cols)

    v8f c[6] = {};                         // index = g*2 + nt
    const int brow[6] = { j0,          j0 + 16,
                          HH + j0,     HH + j0 + 16,
                          2 * HH + j0, 2 * HH + j0 + 16 };
    wmma_gemm_pipe<6, HH / 32>(hb_old, m0, HH, whh, brow, HH, lane, c);

    // Fused gate math + state update. c[g*2+nt][v] is gh for gate g,
    // row = m0 + v + 8*(lane/16), col j = j0 + nt*16 + lane%16.
    const int rbase = m0 + ((lane >> 4) << 3);
#pragma unroll
    for (int nt = 0; nt < 2; ++nt) {
        const int j = j0 + nt * 16 + (lane & 15);
        const float br = bhh[j];
        const float bz = bhh[HH + j];
        const float bn = bhh[2 * HH + j];
#pragma unroll
        for (int v = 0; v < 8; ++v) {
            const int row = rbase + v;
            const size_t gbase = (size_t)row * GG;
            const float ir  = gi_t[gbase + j];
            const float iz  = gi_t[gbase + HH + j];
            const float in_ = gi_t[gbase + 2 * HH + j];
            const float r = sigmoidf_(ir + c[0 * 2 + nt][v] + br);
            const float z = sigmoidf_(iz + c[1 * 2 + nt][v] + bz);
            const float n = tanhf(in_ + r * (c[2 * 2 + nt][v] + bn));
            const size_t hidx = (size_t)row * HH + j;
            const float hnew = (1.0f - z) * n + z * hf_old[hidx];
            hf_new[hidx] = hnew;
            const bf16_t hb = (bf16_t)hnew;
            hb_new[hidx] = hb;
            hs_t[hidx]   = hb;
        }
    }
}

// ---------------------------------------------------------------------------
// preds = hs @ W_dec^T + b_dec; SSE loss vs gt[1:] reduced into out[0];
// preds written to out[1..]. grid (255, 4), block 256.
// ---------------------------------------------------------------------------
__global__ void __launch_bounds__(256) decode_loss_kernel(const bf16_t* __restrict__ hsb,  // [32640,1024]
                                                          const bf16_t* __restrict__ wdec, // [256,1024]
                                                          const float*  __restrict__ bdec, // [256]
                                                          const float*  __restrict__ gt,   // [256,128,256]
                                                          float* __restrict__ out) {
    const int lane = threadIdx.x & 31;
    const int wid  = threadIdx.x >> 5;
    const int m0 = blockIdx.x * 128 + wid * 16;
    const int n0 = blockIdx.y * 64;

    v8f c[4] = {};
    const int brow[4] = { n0, n0 + 16, n0 + 32, n0 + 48 };
    wmma_gemm_pipe<4, HH / 32>(hsb, m0, HH, wdec, brow, HH, lane, c);

    const int rbase = m0 + ((lane >> 4) << 3);
    float lsum = 0.0f;
#pragma unroll
    for (int nt = 0; nt < 4; ++nt) {
        const int col = n0 + nt * 16 + (lane & 15);
        const float bias = bdec[col];
#pragma unroll
        for (int v = 0; v < 8; ++v) {
            const int row = rbase + v;                         // flattened (t,b)
            const float p = c[nt][v] + bias;
            out[1 + (size_t)row * OO + col] = p;               // preds
            const float d = p - gt[(size_t)(row + BB) * OO + col];  // gt[1:]
            lsum += d * d;
        }
    }
    // wave32 reduction, then cross-wave via LDS, one atomic per block.
    for (int o = 16; o > 0; o >>= 1) lsum += __shfl_xor(lsum, o, 32);
    __shared__ float red[8];
    if (lane == 0) red[wid] = lsum;
    __syncthreads();
    if (wid == 0) {
        float s = (lane < 8) ? red[lane] : 0.0f;
        for (int o = 16; o > 0; o >>= 1) s += __shfl_xor(s, o, 32);
        if (lane == 0) atomicAdd(out, s);
    }
}

// ---------------------------------------------------------------------------
// Host-side orchestration
// ---------------------------------------------------------------------------
extern "C" void kernel_launch(void* const* d_in, const int* in_sizes, int n_in,
                              void* d_out, int out_size, void* d_ws, size_t ws_size,
                              hipStream_t stream) {
    const float* x     = (const float*)d_in[0];   // [256,128,512]
    const float* gt    = (const float*)d_in[1];   // [256,128,256]
    const float* W_ih  = (const float*)d_in[2];   // [3072,512]
    const float* W_hh  = (const float*)d_in[3];   // [3072,1024]
    const float* b_ih  = (const float*)d_in[4];   // [3072]
    const float* b_hh  = (const float*)d_in[5];   // [3072]
    const float* W_dec = (const float*)d_in[6];   // [256,1024]
    const float* b_dec = (const float*)d_in[7];   // [256]
    float* out = (float*)d_out;                   // [1 + 255*128*256]

    // Workspace carve-out (all region sizes are multiples of 256 bytes).
    char* ws = (char*)d_ws;
    size_t off = 0;
    bf16_t* xb    = (bf16_t*)(ws + off); off += (size_t)MROWS * II * sizeof(bf16_t);
    bf16_t* wihb  = (bf16_t*)(ws + off); off += (size_t)GG * II * sizeof(bf16_t);
    bf16_t* whhb  = (bf16_t*)(ws + off); off += (size_t)GG * HH * sizeof(bf16_t);
    bf16_t* wdecb = (bf16_t*)(ws + off); off += (size_t)OO * HH * sizeof(bf16_t);
    bf16_t* hb0   = (bf16_t*)(ws + off); off += (size_t)BB * HH * sizeof(bf16_t);
    bf16_t* hb1   = (bf16_t*)(ws + off); off += (size_t)BB * HH * sizeof(bf16_t);
    bf16_t* hsb   = (bf16_t*)(ws + off); off += (size_t)MROWS * HH * sizeof(bf16_t);
    float*  hf0   = (float*) (ws + off); off += (size_t)BB * HH * sizeof(float);
    float*  hf1   = (float*) (ws + off); off += (size_t)BB * HH * sizeof(float);
    float*  gi    = (float*) (ws + off); off += (size_t)MROWS * GG * sizeof(float);

    bf16_t* hb[2] = { hb0, hb1 };
    float*  hf[2] = { hf0, hf1 };

    // 1) Quantize operands to bf16.
    {
        int n;
        n = MROWS * II; cvt_f32_bf16_kernel<<<(n + 255) / 256, 256, 0, stream>>>(x, xb, n);
        n = GG * II;    cvt_f32_bf16_kernel<<<(n + 255) / 256, 256, 0, stream>>>(W_ih, wihb, n);
        n = GG * HH;    cvt_f32_bf16_kernel<<<(n + 255) / 256, 256, 0, stream>>>(W_hh, whhb, n);
        n = OO * HH;    cvt_f32_bf16_kernel<<<(n + 255) / 256, 256, 0, stream>>>(W_dec, wdecb, n);
    }

    // 2) h0 = 0 (ping-pong slot 0), loss = 0.
    init_kernel<<<(BB * HH + 255) / 256, 256, 0, stream>>>(hf0, hb0, out);

    // 3) Batched input projection (big WMMA GEMM, includes +b_ih).
    gi_gemm_kernel<<<dim3(MROWS / 128, GG / 64), 256, 0, stream>>>(xb, wihb, b_ih, gi);

    // 4) Sequential GRU scan: one fused kernel per timestep.
    for (int t = 0; t < TM1; ++t) {
        const int o = t & 1, nw = o ^ 1;
        gru_step_kernel<<<HH / 32, 256, 0, stream>>>(
            hb[o], hf[o], hb[nw], hf[nw], whhb, b_hh,
            gi + (size_t)t * BB * GG,
            hsb + (size_t)t * BB * HH);
    }

    // 5) Decode GEMM + bias + SSE loss.
    decode_loss_kernel<<<dim3(MROWS / 128, OO / 64), 256, 0, stream>>>(
        hsb, wdecb, b_dec, gt, out);

    (void)in_sizes; (void)n_in; (void)out_size; (void)ws_size;
}
